// GraphSAGEBlock_41472204210849
// MI455X (gfx1250) — matrix operand order repaired and checked by reading, not compiled, not run on hardware
//
#include <hip/hip_runtime.h>

#define B_ 8
#define N_ 5000
#define K_ 16
#define DIN_ 128
#define H_ 128
#define M_TOTAL (B_ * N_)      // 40000, divisible by 16
#define TILES (M_TOTAL / 16)   // 2500
#define LDSP 136               // padded row stride (halves) to avoid LDS bank conflicts

typedef __attribute__((ext_vector_type(16))) __bf16 v16bf;
typedef __attribute__((ext_vector_type(8)))  float  v8f;
typedef unsigned short ushort_t;
typedef unsigned int   uint_t;

// float -> bf16 via native converts (backend emits v_cvt_pk_bf16_f32 on gfx1250)
__device__ inline uint_t pkbf(float a, float b) {
  union { __bf16 h[2]; uint_t u; } c;
  c.h[0] = (__bf16)a;
  c.h[1] = (__bf16)b;
  return c.u;
}
__device__ inline ushort_t f2bf1(float a) {
  union { __bf16 h; ushort_t u; } c;
  c.h = (__bf16)a;
  return c.u;
}

union Frag { v16bf bf; uint4 q[2]; };

__device__ inline v8f wmma_bf16(const Frag& a, const Frag& b, v8f c) {
  return __builtin_amdgcn_wmma_f32_16x16x32_bf16(
      /*neg_a=*/false, a.bf, /*neg_b=*/false, b.bf,
      /*c_mod=*/(short)0, c, /*reuse_a=*/false, /*reuse_b=*/false);
}

// A fragment (16x32 bf16) from an LDS tile stored row-major [16][LDSP].
// lanes 0-15: row M=lane, K = {kb..kb+7, kb+16..kb+23}, kb=0; lanes 16-31: kb=8.
__device__ inline Frag load_a(const ushort_t* lds, int kc, int lane) {
  const ushort_t* p = lds + (size_t)(lane & 15) * LDSP + kc * 32 + ((lane & 16) ? 8 : 0);
  Frag f;
  f.q[0] = *(const uint4*)(p);        // K = kb .. kb+7
  f.q[1] = *(const uint4*)(p + 16);   // K = kb+16 .. kb+23
  return f;
}

// B fragment (32x16 bf16) from global bf16 weights row-major W[ncol][128]
// (element (K=k, N=n) = W[n][k]). lanes 0-15: N=lane, K = {kb..kb+7, kb+16..kb+23}, kb=0; lanes 16-31: kb=8.
__device__ inline Frag load_b(const ushort_t* __restrict__ W, int ncol_base, int kc, int lane) {
  const ushort_t* p = W + (size_t)(ncol_base + (lane & 15)) * DIN_ + kc * 32 + ((lane & 16) ? 8 : 0);
  Frag f;
  f.q[0] = *(const uint4*)(p);
  f.q[1] = *(const uint4*)(p + 16);
  return f;
}

// gfx1250 has a hardware v_tanh_f32 transcendental — use it when available.
__device__ inline float tanh_fast(float x) {
#if __has_builtin(__builtin_amdgcn_tanhf)
  return __builtin_amdgcn_tanhf(x);
#else
  x = fminf(fmaxf(x, -15.0f), 15.0f);
  float e = __expf(-2.0f * x);
  return (1.0f - e) / (1.0f + e);
#endif
}
// sigmoid(x) = 0.5*tanh(x/2) + 0.5  (1 TRANS + 2 VALU, no divide)
__device__ inline float sigmoid_fast(float x) {
  return __builtin_fmaf(0.5f, tanh_fast(0.5f * x), 0.5f);
}

__device__ inline v8f splat8(float v) {
  v8f r;
#pragma unroll
  for (int i = 0; i < 8; ++i) r[i] = v;
  return r;
}

__device__ inline uint4 pack8(float4 a, float4 b, float w) {
  uint4 p;
  p.x = pkbf(a.x * w, a.y * w);
  p.y = pkbf(a.z * w, a.w * w);
  p.z = pkbf(b.x * w, b.y * w);
  p.w = pkbf(b.z * w, b.w * w);
  return p;
}

// ---------------- prep: fp32 weights -> bf16 in workspace, fused bias ----------------
__global__ void prep_kernel(const float* __restrict__ Wih, const float* __restrict__ Whh,
                            const float* __restrict__ Wself, const float* __restrict__ Wcomb,
                            const float* __restrict__ bih, const float* __restrict__ bhh,
                            ushort_t* __restrict__ wih_bf, ushort_t* __restrict__ whh_bf,
                            ushort_t* __restrict__ wself_bf, ushort_t* __restrict__ wc1_bf,
                            ushort_t* __restrict__ wc2_bf, float* __restrict__ bsum) {
  int tid = blockIdx.x * blockDim.x + threadIdx.x;
  int stride = gridDim.x * blockDim.x;
  for (int t = tid; t < 4 * H_ * DIN_; t += stride) { wih_bf[t] = f2bf1(Wih[t]); whh_bf[t] = f2bf1(Whh[t]); }
  for (int t = tid; t < H_ * DIN_; t += stride) { wself_bf[t] = f2bf1(Wself[t]); }
  for (int t = tid; t < H_ * H_; t += stride) {
    int n = t >> 7, k = t & 127;
    wc1_bf[t] = f2bf1(Wcomb[n * 2 * H_ + k]);
    wc2_bf[t] = f2bf1(Wcomb[n * 2 * H_ + H_ + k]);
  }
  for (int t = tid; t < 4 * H_; t += stride) bsum[t] = bih[t] + bhh[t];
}

// ---------------- fused gather + LSTM + self/comb GEMMs ----------------
// One workgroup = 16 node-rows. 8 waves; wave w owns h-columns [w*16, w*16+16)
// across all four gates, so c/h updates are per-lane fragment math.
// Weights are loop-invariant B-fragments -> compiler keeps them register-resident
// (VGPRs >256 via s_set_vgpr_msb), steady state is LDS A-loads + WMMA only.
__global__ __launch_bounds__(256, 1) void sage_lstm_kernel(
    const float* __restrict__ x, const int* __restrict__ nidx, const float* __restrict__ nwt,
    const ushort_t* __restrict__ wih, const ushort_t* __restrict__ whh,
    const ushort_t* __restrict__ wself, const ushort_t* __restrict__ wc1,
    const ushort_t* __restrict__ wc2,
    const float* __restrict__ bsum, const float* __restrict__ bself,
    const float* __restrict__ bcomb, float* __restrict__ out) {
  __shared__ ushort_t lds_A[16 * LDSP];  // current timestep activations (bf16)
  __shared__ ushort_t lds_H[16 * LDSP];  // hidden state h (bf16)
  __shared__ ushort_t lds_S[16 * LDSP];  // self features (bf16)

  const int tid  = threadIdx.x;
  const int lane = tid & 31;
  const int wv   = tid >> 5;                  // wave 0..7
  const int wg   = blockIdx.x;                // 16-row tile
  const int colh = wv * 16 + (lane & 15);     // this lane's h-column (0..127)
  const int rowoff = (lane & 16) ? 8 : 0;     // C/D fragment row offset

  // gather assignment: thread t loads 8 features of row t/16
  const int grow = tid >> 4;                  // 0..15
  const int gd   = (tid & 15) * 8;            // feature chunk
  const int m_g  = wg * 16 + grow;
  const int b_g  = m_g / N_;
  const int n_g  = m_g - b_g * N_;

  // zero h
  for (int t = tid; t < 16 * LDSP; t += 256) lds_H[t] = 0;

  // hoist gate biases (b_ih + b_hh) into registers
  float bias_g[4];
#pragma unroll
  for (int g = 0; g < 4; ++g) bias_g[g] = bsum[g * H_ + colh];

  v8f cfrag = splat8(0.0f);

  // prefetch first neighbor row
  {
    int idx0 = nidx[n_g * K_];
    __builtin_prefetch(x + ((size_t)b_g * N_ + idx0) * DIN_ + gd, 0, 1);
  }

  for (int k = 0; k < K_; ++k) {
    // ---- gather + scale neighbor features for timestep k ----
    {
      int   idx = nidx[n_g * K_ + k];
      float w   = nwt[n_g * K_ + k];
      const float* src = x + ((size_t)b_g * N_ + idx) * DIN_ + gd;
      float4 f0 = ((const float4*)src)[0];
      float4 f1 = ((const float4*)src)[1];
      if (k + 1 < K_) {  // hide L2 latency of next step's gather behind this step's WMMAs
        int idxn = nidx[n_g * K_ + k + 1];
        __builtin_prefetch(x + ((size_t)b_g * N_ + idxn) * DIN_ + gd, 0, 1);
      }
      *(uint4*)&lds_A[grow * LDSP + gd] = pack8(f0, f1, w);
    }
    __syncthreads();

    // A fragments shared across the 4 gate tiles
    Frag Ax[4], Ah[4];
#pragma unroll
    for (int kc = 0; kc < 4; ++kc) {
      Ax[kc] = load_a(lds_A, kc, lane);
      Ah[kc] = load_a(lds_H, kc, lane);
    }

    v8f acc[4];
#pragma unroll
    for (int g = 0; g < 4; ++g) {
      v8f a = splat8(bias_g[g]);
#pragma unroll
      for (int kc = 0; kc < 4; ++kc)
        a = wmma_bf16(Ax[kc], load_b(wih, g * H_ + wv * 16, kc, lane), a);
#pragma unroll
      for (int kc = 0; kc < 4; ++kc)
        a = wmma_bf16(Ah[kc], load_b(whh, g * H_ + wv * 16, kc, lane), a);
      acc[g] = a;
    }

    __syncthreads();  // all waves done reading lds_H before rewrite

    // gate nonlinearity + state update (per-lane fragment math, v_tanh based)
#pragma unroll
    for (int r = 0; r < 8; ++r) {
      float iv = sigmoid_fast(acc[0][r]);
      float fv = sigmoid_fast(acc[1][r]);
      float gv = tanh_fast(acc[2][r]);
      float ov = sigmoid_fast(acc[3][r]);
      float cv = __builtin_fmaf(fv, cfrag[r], iv * gv);
      cfrag[r] = cv;
      float hv = ov * tanh_fast(cv);
      lds_H[(r + rowoff) * LDSP + colh] = f2bf1(hv);
    }
  }
  __syncthreads();

  // ---- phase 2a: gather own features for self GEMM ----
  {
    const float* src = x + ((size_t)b_g * N_ + n_g) * DIN_ + gd;
    float4 f0 = ((const float4*)src)[0];
    float4 f1 = ((const float4*)src)[1];
    *(uint4*)&lds_A[grow * LDSP + gd] = pack8(f0, f1, 1.0f);
  }
  __syncthreads();

  // ---- phase 2b: S = x @ Wself^T + bself ----
  {
    v8f a = splat8(bself[colh]);
#pragma unroll
    for (int kc = 0; kc < 4; ++kc)
      a = wmma_bf16(load_a(lds_A, kc, lane), load_b(wself, wv * 16, kc, lane), a);
#pragma unroll
    for (int r = 0; r < 8; ++r)
      lds_S[(r + rowoff) * LDSP + colh] = f2bf1(a[r]);
  }
  __syncthreads();

  // ---- phase 2c: out = relu(S @ Wc1^T + H @ Wc2^T + bcomb) ----
  {
    v8f a = splat8(bcomb[colh]);
#pragma unroll
    for (int kc = 0; kc < 4; ++kc)
      a = wmma_bf16(load_a(lds_S, kc, lane), load_b(wc1, wv * 16, kc, lane), a);
#pragma unroll
    for (int kc = 0; kc < 4; ++kc)
      a = wmma_bf16(load_a(lds_H, kc, lane), load_b(wc2, wv * 16, kc, lane), a);
#pragma unroll
    for (int r = 0; r < 8; ++r)
      out[(size_t)(wg * 16 + r + rowoff) * H_ + colh] = fmaxf(a[r], 0.0f);
  }
}

extern "C" void kernel_launch(void* const* d_in, const int* in_sizes, int n_in,
                              void* d_out, int out_size, void* d_ws, size_t ws_size,
                              hipStream_t stream) {
  const float* x     = (const float*)d_in[0];
  const int*   nidx  = (const int*)d_in[1];
  const float* nwt   = (const float*)d_in[2];
  const float* Wih   = (const float*)d_in[3];
  const float* Whh   = (const float*)d_in[4];
  const float* bih   = (const float*)d_in[5];
  const float* bhh   = (const float*)d_in[6];
  const float* Wself = (const float*)d_in[7];
  const float* bself = (const float*)d_in[8];
  const float* Wcomb = (const float*)d_in[9];
  const float* bcomb = (const float*)d_in[10];

  // workspace layout (bf16 weights + fused bias): 362,496 bytes total
  ushort_t* wih_bf   = (ushort_t*)d_ws;            // 65536 halves
  ushort_t* whh_bf   = wih_bf + 4 * H_ * DIN_;     // 65536
  ushort_t* wself_bf = whh_bf + 4 * H_ * H_;       // 16384
  ushort_t* wc1_bf   = wself_bf + H_ * DIN_;       // 16384
  ushort_t* wc2_bf   = wc1_bf + H_ * H_;           // 16384
  float*    bsum     = (float*)(wc2_bf + H_ * H_); // 512 f32

  prep_kernel<<<128, 256, 0, stream>>>(Wih, Whh, Wself, Wcomb, bih, bhh,
                                       wih_bf, whh_bf, wself_bf, wc1_bf, wc2_bf, bsum);
  sage_lstm_kernel<<<TILES, 256, 0, stream>>>(x, nidx, nwt, wih_bf, whh_bf, wself_bf,
                                              wc1_bf, wc2_bf, bsum, bself, bcomb,
                                              (float*)d_out);
}
